// GraphNN_31980326486286
// MI455X (gfx1250) — matrix operand order repaired and checked by reading, not compiled, not run on hardware
//
#include <hip/hip_runtime.h>

// ---------------------------------------------------------------------------
// GCN + RK4 ODE block for MI455X (gfx1250, wave32, WMMA).
// GEMMs run on v_wmma_f32_16x16x32_f16 (f32 accumulate). Edge aggregation is
// L2-resident scatter-add (all feature maps fit in the 192MB L2).
// ---------------------------------------------------------------------------

typedef __attribute__((ext_vector_type(16))) _Float16 v16h;
typedef __attribute__((ext_vector_type(8)))  _Float16 half8;
typedef __attribute__((ext_vector_type(8)))  float    v8f;

#define N_NODES 50000
#define N_EDGES 800000
#define NINP    128
#define D2      256

#define WMMA_F16(a, b, c) \
  __builtin_amdgcn_wmma_f32_16x16x32_f16(false, (a), false, (b), (short)0, (c), false, false)

#define CAT16(lo, hi) \
  __builtin_shufflevector((lo), (hi), 0,1,2,3,4,5,6,7,8,9,10,11,12,13,14,15)

// ---------------------------------------------------------------- utilities
__global__ void k_fill(float* __restrict__ p, float v, int n) {
  int i = blockIdx.x * blockDim.x + threadIdx.x;
  if (i < n) p[i] = v;
}

__global__ void k_deg(const long long* __restrict__ dst, float* __restrict__ deg, int e) {
  int i = blockIdx.x * blockDim.x + threadIdx.x;
  if (i < e) atomicAdd(&deg[(int)dst[i]], 1.0f);
}

__global__ void k_rsqrt(float* __restrict__ d, int n) {
  int i = blockIdx.x * blockDim.x + threadIdx.x;
  if (i < n) d[i] = rsqrtf(d[i]);
}

__global__ void k_f2h(const float* __restrict__ x, _Float16* __restrict__ xh, int n) {
  int i = blockIdx.x * blockDim.x + threadIdx.x;
  if (i < n) xh[i] = (_Float16)x[i];
}

// W is [K,M] row-major fp32; produce WT [M,K] fp16 so B-fragment loads are
// contiguous along K (global_load_b128 friendly).
__global__ void k_wT2h(const float* __restrict__ W, _Float16* __restrict__ WT, int K, int M) {
  int i = blockIdx.x * blockDim.x + threadIdx.x;
  if (i < K * M) {
    int k = i / M, m = i % M;
    WT[(size_t)m * K + k] = (_Float16)W[i];
  }
}

// out = a + s*b
__global__ void k_axpb(const float* __restrict__ a, const float* __restrict__ b,
                       float s, float* __restrict__ out, int n) {
  int i = blockIdx.x * blockDim.x + threadIdx.x;
  if (i < n) out[i] = a[i] + s * b[i];
}

// ------------------------------------------------------------------- GEMM
// Y[nrows,M] = Xh[nrows,K] * WhT[M,K]^T  (fp16 in, fp32 out).
// One wave computes a 16x64 strip: 1 A fragment reused over 4 B fragments.
// C/D layout per CDNA5 ISA 7.12.2; A 16x32 / B 32x16 use contiguous k (lane half h: k 16h..16h+15).
__global__ __launch_bounds__(256)
void k_gemm(const _Float16* __restrict__ Xh, const _Float16* __restrict__ WhT,
            float* __restrict__ Y, int nrows, int K, int M) {
  const int wave = blockIdx.x * (blockDim.x >> 5) + (threadIdx.x >> 5);
  const int lane = threadIdx.x & 31;
  const int strips = M >> 6;                 // 64-column strips
  const int row0 = (wave / strips) << 4;
  const int col0 = (wave % strips) << 6;
  if (row0 >= nrows) return;

  const int m  = lane & 15;                  // row (A) / column (B,C) within tile
  const int hi = lane >> 4;                  // half-wave select
  const _Float16* arow  = Xh  + (size_t)(row0 + m) * K + hi * 16;
  const _Float16* bbase = WhT + (size_t)(col0 + m) * K + hi * 16;
  const size_t tstride = (size_t)16 * K;     // 16 columns per B tile

  v8f acc0 = {}, acc1 = {}, acc2 = {}, acc3 = {};
  for (int kb = 0; kb < K; kb += 32) {
    // A 16x32: lanes 0-15 hold K[0..15] of row m; lanes 16-31 K[16..31]
    half8 alo = *(const half8*)(arow + kb);
    half8 ahi = *(const half8*)(arow + kb + 8);
    v16h a = CAT16(alo, ahi);

    // B 32x16 per tile: lanes 0-15 hold K[0..15] of column m; lanes 16-31 K[16..31]
    const _Float16* bp0 = bbase + kb;
    v16h b0 = CAT16(*(const half8*)(bp0),                *(const half8*)(bp0 + 8));
    v16h b1 = CAT16(*(const half8*)(bp0 + tstride),      *(const half8*)(bp0 + tstride + 8));
    v16h b2 = CAT16(*(const half8*)(bp0 + 2 * tstride),  *(const half8*)(bp0 + 2 * tstride + 8));
    v16h b3 = CAT16(*(const half8*)(bp0 + 3 * tstride),  *(const half8*)(bp0 + 3 * tstride + 8));

    acc0 = WMMA_F16(a, b0, acc0);
    acc1 = WMMA_F16(a, b1, acc1);
    acc2 = WMMA_F16(a, b2, acc2);
    acc3 = WMMA_F16(a, b3, acc3);
  }

  // C/D layout: lanes 0-15 => N=lane, VGPR r => M=r; lanes 16-31 => M=8+r.
  float* y0 = Y + (size_t)(row0 + hi * 8) * M + col0 + m;
#pragma unroll
  for (int r = 0; r < 8; ++r) {
    y0[(size_t)r * M +  0] = acc0[r];
    y0[(size_t)r * M + 16] = acc1[r];
    y0[(size_t)r * M + 32] = acc2[r];
    y0[(size_t)r * M + 48] = acc3[r];
  }
}

// --------------------------------------------------- conv: self-loop + bias
__global__ void k_selfinit(const float* __restrict__ h, const float* __restrict__ dinv,
                           const float* __restrict__ bias, float* __restrict__ agg,
                           int n, int dshift) {
  int i = blockIdx.x * blockDim.x + threadIdx.x;
  int total = n << dshift;
  if (i < total) {
    int node = i >> dshift;
    int d = i & ((1 << dshift) - 1);
    float di = dinv[node];
    agg[i] = di * di * h[i] + bias[d];
  }
}

// ------------------------------------------------------- edge scatter-add
// One wave per edge: coalesced 128B gathers of h[src], L2-resident atomics.
__global__ __launch_bounds__(256)
void k_edge(const long long* __restrict__ src, const long long* __restrict__ dst,
            const float* __restrict__ dinv, const float* __restrict__ h,
            float* __restrict__ agg, int e, int D) {
  int wave = blockIdx.x * (blockDim.x >> 5) + (threadIdx.x >> 5);
  int lane = threadIdx.x & 31;
  if (wave >= e) return;
  int s = (int)src[wave];
  int t = (int)dst[wave];
  float nrm = dinv[s] * dinv[t];
  const float* hs = h + (size_t)s * D;
  float* at = agg + (size_t)t * D;
  for (int j = lane; j < D; j += 32) atomicAdd(&at[j], nrm * hs[j]);
}

// --------------------------------------------------------------- l2 norm
__device__ __forceinline__ float wave_sum(float s) {
#pragma unroll
  for (int m = 16; m >= 1; m >>= 1) s += __shfl_xor(s, m, 32);
  return s;
}

__global__ __launch_bounds__(256)
void k_l2n_f32(const float* __restrict__ in, float* __restrict__ out, int n, int D) {
  int node = blockIdx.x * (blockDim.x >> 5) + (threadIdx.x >> 5);
  int lane = threadIdx.x & 31;
  if (node >= n) return;
  const float* p = in + (size_t)node * D;
  float s = 0.f;
  for (int j = lane; j < D; j += 32) { float v = p[j]; s += v * v; }
  s = wave_sum(s);
  float scale = 1.0f / fmaxf(sqrtf(s), 1e-12f);
  float* o = out + (size_t)node * D;
  for (int j = lane; j < D; j += 32) o[j] = p[j] * scale;
}

// l2norm fused with fp16 conversion (feeds the next GEMM directly).
__global__ __launch_bounds__(256)
void k_l2n_f16(const float* __restrict__ in, _Float16* __restrict__ out, int n, int D) {
  int node = blockIdx.x * (blockDim.x >> 5) + (threadIdx.x >> 5);
  int lane = threadIdx.x & 31;
  if (node >= n) return;
  const float* p = in + (size_t)node * D;
  float s = 0.f;
  for (int j = lane; j < D; j += 32) { float v = p[j]; s += v * v; }
  s = wave_sum(s);
  float scale = 1.0f / fmaxf(sqrtf(s), 1e-12f);
  _Float16* o = out + (size_t)node * D;
  for (int j = lane; j < D; j += 32) o[j] = (_Float16)(p[j] * scale);
}

// ---------------------------------------------------------------------------
extern "C" void kernel_launch(void* const* d_in, const int* in_sizes, int n_in,
                              void* d_out, int out_size, void* d_ws, size_t ws_size,
                              hipStream_t stream) {
  (void)in_sizes; (void)n_in; (void)out_size; (void)ws_size;

  const float*     x0   = (const float*)d_in[0];      // [N,128]
  const long long* eidx = (const long long*)d_in[1];  // [2,E] int64
  const float* W1  = (const float*)d_in[2];  const float* b1  = (const float*)d_in[3];
  const float* Wf1 = (const float*)d_in[4];  const float* bf1 = (const float*)d_in[5];
  const float* Wf2 = (const float*)d_in[6];  const float* bf2 = (const float*)d_in[7];
  const float* W2  = (const float*)d_in[8];  const float* b2  = (const float*)d_in[9];
  const long long* esrc = eidx;
  const long long* edst = eidx + N_EDGES;
  float* out = (float*)d_out;                         // [N,128]

  // ---- workspace carve-up (256B aligned) ----
  char* w = (char*)d_ws;
  auto take = [&](size_t bytes) -> void* {
    void* p = (void*)w;
    w += (bytes + 255) & ~(size_t)255;
    return p;
  };
  float*    dinv = (float*)take((size_t)N_NODES * 4);
  _Float16* W1h  = (_Float16*)take((size_t)NINP * D2 * 2);   // [256,128]
  _Float16* Wf1h = (_Float16*)take((size_t)D2 * D2 * 2);     // [256,256]
  _Float16* Wf2h = (_Float16*)take((size_t)D2 * D2 * 2);     // [256,256]
  _Float16* W2h  = (_Float16*)take((size_t)D2 * NINP * 2);   // [128,256]
  _Float16* Xh   = (_Float16*)take((size_t)N_NODES * D2 * 2);
  float* hb  = (float*)take((size_t)N_NODES * D2 * 4);
  float* agg = (float*)take((size_t)N_NODES * D2 * 4);
  float* x   = (float*)take((size_t)N_NODES * D2 * 4);
  float* z   = (float*)take((size_t)N_NODES * D2 * 4);
  float* kb  = (float*)take((size_t)N_NODES * D2 * 4);
  float* acc = (float*)take((size_t)N_NODES * D2 * 4);

  const int T = 256;
  auto grid1 = [&](int n) { return dim3((unsigned)((n + T - 1) / T)); };

  // ---- degrees / symmetric norm ----
  k_fill<<<grid1(N_NODES), T, 0, stream>>>(dinv, 1.0f, N_NODES);          // self-loop
  k_deg<<<grid1(N_EDGES), T, 0, stream>>>(edst, dinv, N_EDGES);
  k_rsqrt<<<grid1(N_NODES), T, 0, stream>>>(dinv, N_NODES);

  // ---- weight prep: fp16, transposed [M,K] ----
  k_wT2h<<<grid1(NINP * D2), T, 0, stream>>>(W1, W1h, NINP, D2);
  k_wT2h<<<grid1(D2 * D2),   T, 0, stream>>>(Wf1, Wf1h, D2, D2);
  k_wT2h<<<grid1(D2 * D2),   T, 0, stream>>>(Wf2, Wf2h, D2, D2);
  k_wT2h<<<grid1(D2 * NINP), T, 0, stream>>>(W2, W2h, D2, NINP);

  auto gemm = [&](const _Float16* Wh, float* Y, int K, int M) {
    int waves = (N_NODES / 16) * (M / 64);
    k_gemm<<<dim3((unsigned)((waves + 7) / 8)), T, 0, stream>>>(Xh, Wh, Y, N_NODES, K, M);
  };
  auto conv_post = [&](const float* bias, float* aggp, int D) {
    int dshift = (D == D2) ? 8 : 7;
    k_selfinit<<<grid1(N_NODES * D), T, 0, stream>>>(hb, dinv, bias, aggp, N_NODES, dshift);
    k_edge<<<dim3((unsigned)((N_EDGES + 7) / 8)), T, 0, stream>>>(esrc, edst, dinv, hb, aggp,
                                                                  N_EDGES, D);
  };
  auto axpb = [&](const float* a, const float* b, float s, float* o) {
    k_axpb<<<grid1(N_NODES * D2), T, 0, stream>>>(a, b, s, o, N_NODES * D2);
  };
  dim3 gwave((unsigned)((N_NODES + 7) / 8));

  // ---- conv1: x = l2norm(gcn(x0, W1, b1)) ----
  k_f2h<<<grid1(N_NODES * NINP), T, 0, stream>>>(x0, Xh, N_NODES * NINP);
  gemm(W1h, hb, NINP, D2);
  conv_post(b1, agg, D2);
  k_l2n_f32<<<gwave, T, 0, stream>>>(agg, x, N_NODES, D2);

  // ---- f(z): agg = gcn(l2norm(gcn(z,Wf1,bf1)), Wf2, bf2); k = agg + z ----
  auto f_eval = [&](const float* zin) {
    k_f2h<<<grid1(N_NODES * D2), T, 0, stream>>>(zin, Xh, N_NODES * D2);
    gemm(Wf1h, hb, D2, D2);
    conv_post(bf1, agg, D2);
    k_l2n_f16<<<gwave, T, 0, stream>>>(agg, Xh, N_NODES, D2);   // fused norm + fp16
    gemm(Wf2h, hb, D2, D2);
    conv_post(bf2, agg, D2);
  };

  // ---- RK4, h = 1 ----
  f_eval(x);                      // k1
  axpb(agg, x, 1.0f, acc);        // acc = k1
  axpb(x, acc, 0.5f, z);          // z = x + 0.5 k1
  f_eval(z);                      // k2
  axpb(agg, z, 1.0f, kb);
  axpb(acc, kb, 2.0f, acc);       // acc += 2 k2
  axpb(x, kb, 0.5f, z);           // z = x + 0.5 k2
  f_eval(z);                      // k3
  axpb(agg, z, 1.0f, kb);
  axpb(acc, kb, 2.0f, acc);       // acc += 2 k3
  axpb(x, kb, 1.0f, z);           // z = x + k3
  f_eval(z);                      // k4
  axpb(agg, z, 1.0f, kb);
  axpb(acc, kb, 1.0f, acc);       // acc += k4
  axpb(x, acc, 1.0f / 6.0f, x);   // x += acc/6

  // ---- final conv -> d_out ([N,128], fully written by selfinit first) ----
  k_f2h<<<grid1(N_NODES * D2), T, 0, stream>>>(x, Xh, N_NODES * D2);
  gemm(W2h, hb, D2, NINP);
  conv_post(b2, out, NINP);
}
